// MultiHeadAttention_747324310354
// MI455X (gfx1250) — matrix operand order repaired and checked by reading, not compiled, not run on hardware
//
#include <hip/hip_runtime.h>
#include <stdint.h>

// ---------------- WMMA types (gfx1250, wave32) ----------------
typedef __attribute__((ext_vector_type(16))) __bf16 v16bf;
typedef __attribute__((ext_vector_type(8)))  float  v8f;
typedef __attribute__((ext_vector_type(4)))  unsigned int v4u;

union V16 {
  v16bf v;
  v4u   q[2];
  unsigned short s[16];
};

__device__ __forceinline__ unsigned short f2bf(float f) {
  union { float f; unsigned int u; } x; x.f = f;
  unsigned int u = x.u;
  u += 0x7FFFu + ((u >> 16) & 1u);   // round-to-nearest-even
  return (unsigned short)(u >> 16);
}

// ---------------- problem constants ----------------
#define BATCH   8
#define TQ      512
#define SL      2048
#define HID     1024
#define NHEAD   16
#define DK      64
// score buffer pitch (f32 words): 2060 % 64 == 12 -> WMMA tile stores (rows r, r+8)
// and softmax row-groups (4 consecutive rows/wave) hit disjoint bank spans.
#define SPITCH  2060
// prob buffer pitch (bf16 elems): 2056*2B = 1028 words; 1028 % 64 == 4 -> each of the
// 16 lanes' b128 A-operand loads covers a distinct 4-bank span (conflict-free).
#define PPITCH  2056

// ---------------- convert kernels ----------------
__global__ void cvt_rows_kernel(const float* __restrict__ src,
                                unsigned short* __restrict__ dst, int n) {
  int i = blockIdx.x * blockDim.x + threadIdx.x;
  int stride = gridDim.x * blockDim.x;
  for (; i < n; i += stride) dst[i] = f2bf(src[i]);
}

// W[K][N] (row-major) -> Wt[N][K] bf16, so B-operand loads are K-contiguous
__global__ void cvt_transpose_kernel(const float* __restrict__ W,
                                     unsigned short* __restrict__ Wt,
                                     int K, int N) {
  int i = blockIdx.x * blockDim.x + threadIdx.x;
  int stride = gridDim.x * blockDim.x;
  int total = K * N;
  for (; i < total; i += stride) {
    int k = i / N, n = i - k * N;
    Wt[(size_t)n * K + k] = f2bf(W[i]);
  }
}

// ---------------- generic bf16 WMMA GEMM ----------------
// C[M][N] = A[M][K] * Wt[N][K]^T (+bias). One wave per 16x64 tile.
// out_mode: 0 = f32 row-major, 1 = bf16 row-major, 2 = bf16 transposed [col][M]
#define LOAD_A(dst, off)                                   \
  do {                                                     \
    const unsigned short* _p = aBase + (off);              \
    (dst).q[0] = *(const v4u*)(_p);                        \
    (dst).q[1] = *(const v4u*)(_p + 16);                   \
  } while (0)
#define LOAD_B4(dst, off)                                  \
  do {                                                     \
    _Pragma("unroll")                                      \
    for (int _nt = 0; _nt < 4; ++_nt) {                    \
      const unsigned short* _p = bBase[_nt] + (off);       \
      (dst)[_nt].q[0] = *(const v4u*)(_p);                 \
      (dst)[_nt].q[1] = *(const v4u*)(_p + 16);            \
    }                                                      \
  } while (0)
#define WMMA4(cacc, am, bm)                                              \
  do {                                                                   \
    _Pragma("unroll")                                                    \
    for (int _nt = 0; _nt < 4; ++_nt)                                    \
      (cacc)[_nt] = __builtin_amdgcn_wmma_f32_16x16x32_bf16(             \
          false, (am).v, false, (bm)[_nt].v, (short)0, (cacc)[_nt],      \
          false, false);                                                 \
  } while (0)

__global__ __launch_bounds__(32)
void gemm_bf16_kernel(const unsigned short* __restrict__ A,
                      const unsigned short* __restrict__ Bt,
                      const float* __restrict__ bias,       // may be null
                      unsigned short* __restrict__ outB,
                      float* __restrict__ outF,
                      int M, int N, int K, int out_mode) {
  const int lane = threadIdx.x & 31;
  const int half = lane >> 4;   // lane group 0/1
  const int l16  = lane & 15;
  const int mtile = blockIdx.x;
  const int ng    = blockIdx.y; // group of 64 output columns

  // A layout: lane M=l16; halves j: K = (half?8:0) + (j>=8?16:0) + (j&7) + k0
  const unsigned short* aBase = A + (size_t)(mtile * 16 + l16) * K + half * 8;
  // B layout: lane N=l16; halves j: K = (half?8:0) + (j>=8?16:0) + (j&7) + k0 (in Wt)
  const unsigned short* bBase[4];
#pragma unroll
  for (int nt = 0; nt < 4; ++nt)
    bBase[nt] = Bt + (size_t)(ng * 64 + nt * 16 + l16) * K + half * 8;

  v8f c[4] = {};
  V16 a0, a1, b0[4], b1[4];

  LOAD_A(a0, 0);
  LOAD_B4(b0, 0);
  for (int k0 = 0; k0 < K; k0 += 64) {
    // prefetch k0+32 while computing k0
    LOAD_A(a1, k0 + 32);
    LOAD_B4(b1, k0 + 32);
    WMMA4(c, a0, b0);
    // prefetch k0+64 while computing k0+32
    if (k0 + 64 < K) {
      LOAD_A(a0, k0 + 64);
      LOAD_B4(b0, k0 + 64);
    }
    WMMA4(c, a1, b1);
  }

  if (out_mode == 2) {
    // bf16 transposed [col][M]: 8 contiguous rows per lane -> single b128 store
#pragma unroll
    for (int nt = 0; nt < 4; ++nt) {
      int col = ng * 64 + nt * 16 + l16;
      V16 pk;
#pragma unroll
      for (int r = 0; r < 8; ++r) pk.s[r] = f2bf(c[nt][r]);
      *(v4u*)(outB + (size_t)col * M + mtile * 16 + half * 8) = pk.q[0];
    }
  } else {
#pragma unroll
    for (int nt = 0; nt < 4; ++nt) {
      int col = ng * 64 + nt * 16 + l16;
      float bv = bias ? bias[col] : 0.f;
#pragma unroll
      for (int r = 0; r < 8; ++r) {
        int row = mtile * 16 + r + half * 8;  // C/D layout
        float val = c[nt][r] + bv;
        if (out_mode == 1) outB[(size_t)row * N + col] = f2bf(val);
        else               outF[(size_t)row * N + col] = val;
      }
    }
  }
}

// ---------------- fused attention core ----------------
// One workgroup (4 waves) per (batch, 16-row t-tile); loops over 16 heads.
// Per-head mean accumulated in the block-owned (L2-resident) output region.
__global__ __launch_bounds__(128)
void attn_kernel(const unsigned short* __restrict__ Qb,   // [B*TQ][HID] bf16
                 const unsigned short* __restrict__ Kb,   // [B*SL][HID] bf16
                 const unsigned short* __restrict__ Vt,   // [HID][B*SL] bf16 (transposed)
                 const int* __restrict__ mask,            // [B][SL]
                 unsigned short* __restrict__ concat,     // [B*TQ][HID] bf16
                 float* __restrict__ outMean) {           // [B*TQ][SL] f32
  extern __shared__ char smem_raw[];
  float* Sb = (float*)smem_raw;                                  // [16][SPITCH] f32 scores
  unsigned short* Pb = (unsigned short*)(smem_raw + 16 * SPITCH * 4); // [16][PPITCH] bf16 probs

  const int tid  = threadIdx.x;
  const int lane = tid & 31;
  const int wave = tid >> 5;        // 0..3
  const int half = lane >> 4;
  const int l16  = lane & 15;
  const int ttile = blockIdx.x;     // 0..31
  const int b     = blockIdx.y;     // 0..7
  const int qrow0 = b * TQ + ttile * 16;
  const int krow0 = b * SL;

  for (int h = 0; h < NHEAD; ++h) {
    const int hoff = h * DK;

    // Q tile A-operands for the two 32-wide K chunks of d_k=64
    V16 aq0, aq1;
    {
      const unsigned short* ap = Qb + (size_t)(qrow0 + l16) * HID + hoff + half * 8;
      aq0.q[0] = *(const v4u*)(ap);       aq0.q[1] = *(const v4u*)(ap + 16);
      aq1.q[0] = *(const v4u*)(ap + 32);  aq1.q[1] = *(const v4u*)(ap + 48);
    }

    // ---- S = Q K^T : waves split the 128 column tiles; double-buffered ----
    const unsigned short* kBase =
        Kb + (size_t)(krow0 + l16) * HID + hoff + half * 8;
#define LOAD_K(d0, d1, jt)                                            \
    do {                                                              \
      const unsigned short* _p = kBase + (size_t)(jt) * 16 * HID;     \
      (d0).q[0] = *(const v4u*)(_p);      (d0).q[1] = *(const v4u*)(_p + 16); \
      (d1).q[0] = *(const v4u*)(_p + 32); (d1).q[1] = *(const v4u*)(_p + 48); \
    } while (0)
#define QK_TILE(bk0, bk1, jt)                                                  \
    do {                                                                       \
      v8f _c = {};                                                             \
      _c = __builtin_amdgcn_wmma_f32_16x16x32_bf16(false, aq0.v, false,        \
              (bk0).v, (short)0, _c, false, false);                            \
      _c = __builtin_amdgcn_wmma_f32_16x16x32_bf16(false, aq1.v, false,        \
              (bk1).v, (short)0, _c, false, false);                            \
      _Pragma("unroll")                                                        \
      for (int _r = 0; _r < 8; ++_r)                                           \
        Sb[(_r + half * 8) * SPITCH + (jt) * 16 + l16] = _c[_r];               \
    } while (0)
    {
      V16 k0a, k0b, k1a, k1b;
      LOAD_K(k0a, k0b, wave);
      for (int jt = wave; jt < SL / 16; jt += 8) {
        LOAD_K(k1a, k1b, jt + 4);                      // always valid (max 124+... <128)
        QK_TILE(k0a, k0b, jt);
        if (jt + 8 < SL / 16) LOAD_K(k0a, k0b, jt + 8);
        QK_TILE(k1a, k1b, jt + 4);
      }
    }
    __syncthreads();

    // ---- masked softmax over s; bf16 probs -> Pb; mean -> global (L2) ----
    {
      const int row = tid >> 3;
      const int sub = tid & 7;
      const int* mrow = mask + b * SL;
      float* Srow = Sb + row * SPITCH;
      float m = -3.0e38f;
      for (int cc = sub; cc < SL; cc += 8) {
        float v = Srow[cc] * 0.125f + (mrow[cc] ? 0.f : -1.0e20f);
        Srow[cc] = v;
        m = fmaxf(m, v);
      }
      m = fmaxf(m, __shfl_xor(m, 1, 8));
      m = fmaxf(m, __shfl_xor(m, 2, 8));
      m = fmaxf(m, __shfl_xor(m, 4, 8));
      float ssum = 0.f;
      for (int cc = sub; cc < SL; cc += 8) {
        float e = __expf(Srow[cc] - m);
        Srow[cc] = e;
        ssum += e;
      }
      ssum += __shfl_xor(ssum, 1, 8);
      ssum += __shfl_xor(ssum, 2, 8);
      ssum += __shfl_xor(ssum, 4, 8);
      float inv = 1.f / ssum;
      unsigned short* Prow = Pb + row * PPITCH;
      float* gm = outMean + (size_t)(qrow0 + row) * SL;
      for (int cc = sub; cc < SL; cc += 8) {
        float p = Srow[cc] * inv;
        Prow[cc] = f2bf(p);
        float prev = (h == 0) ? 0.f : gm[cc];
        gm[cc] = prev + p * (1.0f / NHEAD);
      }
    }
    __syncthreads();

    // ---- O_tile = P V : each wave owns one 16-col slice; double-buffered ----
    {
      const int nt = wave;  // 0..3
      // A operand: bf16 probs, contiguous in Pb per the A layout
      const unsigned short* pBase = Pb + l16 * PPITCH + half * 8;
      // B operand: Vt[d][b*SL + s], contiguous along s
      const unsigned short* vBase =
          Vt + (size_t)(hoff + nt * 16 + l16) * (BATCH * SL) + krow0 + half * 8;
#define LOAD_P(dst, kc)                                               \
      do {                                                            \
        const unsigned short* _p = pBase + (kc) * 32;                 \
        (dst).q[0] = *(const v4u*)(_p);                               \
        (dst).q[1] = *(const v4u*)(_p + 16);                          \
      } while (0)
#define LOAD_V(dst, kc)                                               \
      do {                                                            \
        const unsigned short* _p = vBase + (kc) * 32;                 \
        (dst).q[0] = *(const v4u*)(_p);                               \
        (dst).q[1] = *(const v4u*)(_p + 16);                          \
      } while (0)
      v8f c = {};
      V16 p0, p1, v0, v1;
      LOAD_P(p0, 0);
      LOAD_V(v0, 0);
      for (int kc = 0; kc < SL / 32; kc += 2) {
        LOAD_P(p1, kc + 1);                           // kc+1 <= 63 always
        LOAD_V(v1, kc + 1);
        c = __builtin_amdgcn_wmma_f32_16x16x32_bf16(
            false, p0.v, false, v0.v, (short)0, c, false, false);
        if (kc + 2 < SL / 32) { LOAD_P(p0, kc + 2); LOAD_V(v0, kc + 2); }
        c = __builtin_amdgcn_wmma_f32_16x16x32_bf16(
            false, p1.v, false, v1.v, (short)0, c, false, false);
      }
#pragma unroll
      for (int r = 0; r < 8; ++r) {
        int grow = qrow0 + r + half * 8;
        concat[(size_t)grow * HID + hoff + nt * 16 + l16] = f2bf(c[r]);
      }
    }
    __syncthreads();
  }
}

// ---------------- host launcher ----------------
extern "C" void kernel_launch(void* const* d_in, const int* in_sizes, int n_in,
                              void* d_out, int out_size, void* d_ws, size_t ws_size,
                              hipStream_t stream) {
  const float* hiddens = (const float*)d_in[0];   // [8,2048,1024]
  const float* qh      = (const float*)d_in[1];   // [8,512,1024]
  const int*   mask    = (const int*)d_in[2];     // [8,2048]
  const float* W_q     = (const float*)d_in[3];
  const float* b_q     = (const float*)d_in[4];
  const float* W_k     = (const float*)d_in[5];
  const float* W_v     = (const float*)d_in[6];
  const float* W_o     = (const float*)d_in[7];
  const float* b_o     = (const float*)d_in[8];
  float* out = (float*)d_out;

  // workspace layout (bytes)
  char* ws = (char*)d_ws;
  const size_t WSZ  = (size_t)HID * HID * sizeof(unsigned short);     // 2 MB
  unsigned short* Wq_t = (unsigned short*)(ws + 0 * WSZ);
  unsigned short* Wk_t = (unsigned short*)(ws + 1 * WSZ);
  unsigned short* Wv_t = (unsigned short*)(ws + 2 * WSZ);
  unsigned short* Wo_t = (unsigned short*)(ws + 3 * WSZ);
  size_t off = 4 * WSZ;
  const size_t XQ_N = (size_t)BATCH * TQ * HID;   // 4.19M elems
  const size_t XH_N = (size_t)BATCH * SL * HID;   // 16.8M elems
  unsigned short* Xq = (unsigned short*)(ws + off); off += XQ_N * 2;
  unsigned short* Xh = (unsigned short*)(ws + off); off += XH_N * 2;
  unsigned short* Qb = (unsigned short*)(ws + off); off += XQ_N * 2;
  unsigned short* Kb = (unsigned short*)(ws + off); off += XH_N * 2;
  unsigned short* Vt = (unsigned short*)(ws + off); off += XH_N * 2;  // [HID][B*SL]
  unsigned short* Cc = (unsigned short*)(ws + off); off += XQ_N * 2;  // concat

  // 1) weight transposes + activation conversion to bf16
  cvt_transpose_kernel<<<dim3(1024), dim3(256), 0, stream>>>(W_q, Wq_t, HID, HID);
  cvt_transpose_kernel<<<dim3(1024), dim3(256), 0, stream>>>(W_k, Wk_t, HID, HID);
  cvt_transpose_kernel<<<dim3(1024), dim3(256), 0, stream>>>(W_v, Wv_t, HID, HID);
  cvt_transpose_kernel<<<dim3(1024), dim3(256), 0, stream>>>(W_o, Wo_t, HID, HID);
  cvt_rows_kernel<<<dim3(4096), dim3(256), 0, stream>>>(qh,      Xq, (int)XQ_N);
  cvt_rows_kernel<<<dim3(8192), dim3(256), 0, stream>>>(hiddens, Xh, (int)XH_N);

  // 2) projections (bf16 WMMA, f32 accumulate); V emitted transposed
  gemm_bf16_kernel<<<dim3(BATCH * TQ / 16, HID / 64), dim3(32), 0, stream>>>(
      Xq, Wq_t, b_q, Qb, nullptr, BATCH * TQ, HID, HID, 1);
  gemm_bf16_kernel<<<dim3(BATCH * SL / 16, HID / 64), dim3(32), 0, stream>>>(
      Xh, Wk_t, nullptr, Kb, nullptr, BATCH * SL, HID, HID, 1);
  gemm_bf16_kernel<<<dim3(BATCH * SL / 16, HID / 64), dim3(32), 0, stream>>>(
      Xh, Wv_t, nullptr, Vt, nullptr, BATCH * SL, HID, HID, 2);

  // 3) fused attention (scores + softmax + mean + P*V), ~193 KB dynamic LDS
  const size_t lds = (size_t)16 * SPITCH * 4 + (size_t)16 * PPITCH * 2;
  attn_kernel<<<dim3(TQ / 16, BATCH), dim3(128), lds, stream>>>(
      Qb, Kb, Vt, mask, Cc, out + (size_t)BATCH * TQ * HID);

  // 4) output projection -> d_out[0 : B*TQ*HID]
  gemm_bf16_kernel<<<dim3(BATCH * TQ / 16, HID / 64), dim3(32), 0, stream>>>(
      Cc, Wo_t, b_o, nullptr, out, BATCH * TQ, HID, HID, 0);
}
